// CPTAttention_86629490360408
// MI455X (gfx1250) — matrix operand order repaired and checked by reading, not compiled, not run on hardware
//
#include <hip/hip_runtime.h>
#include <hip/hip_bf16.h>
#include <math.h>

// ---------------------------------------------------------------------------
// Types for CDNA5 WMMA (wave32): bf16 16x16x32, f32 accumulate.
// ---------------------------------------------------------------------------
typedef __bf16 bf16_t;
typedef __attribute__((ext_vector_type(16))) __bf16 v16bf;
typedef __attribute__((ext_vector_type(8)))  __bf16 v8bf;
typedef __attribute__((ext_vector_type(8)))  float  v8f;

#define WMMA_BF16(A, B, C) \
  __builtin_amdgcn_wmma_f32_16x16x32_bf16(false, (A), false, (B), (short)0, (C), false, false)

// Problem dims
#define BATCH  2
#define SEQ    2048
#define CDIM   1024
#define NHEAD  16
#define HDIM   64
#define RANK   16
#define MROWS  (BATCH * SEQ)   // 4096
#define NQKV   (3 * CDIM)      // 3072

static __device__ inline v8f vzero8() {
  v8f z;
#pragma unroll
  for (int i = 0; i < 8; ++i) z[i] = 0.0f;
  return z;
}

static __device__ inline v16bf pack16(v8bf lo, v8bf hi) {
  v16bf r;
#pragma unroll
  for (int i = 0; i < 8; ++i) { r[i] = lo[i]; r[i + 8] = hi[i]; }
  return r;
}

// A operand (16x32 bf16) from row-major [M][K]:
// lane L: row = row0 + (L&15); elems 0..7 = K k0+8h.., elems 8..15 = K k0+16+8h..
static __device__ inline v16bf load_a16x32(const bf16_t* base, int ld, int row0, int k0, int lane) {
  const int m  = lane & 15;
  const int hi = lane >> 4;
  const bf16_t* p = base + (size_t)(row0 + m) * ld + k0 + 8 * hi;
  v8bf x0 = *(const v8bf*)(p);
  v8bf x1 = *(const v8bf*)(p + 16);
  return pack16(x0, x1);
}

// B operand (32x16 bf16) from [N][K] row-major storage (operand column n = stored row n):
// lane L: col = col0 + (L&15); elems 0..7 = K k0+8h.., elems 8..15 = K k0+16+8h..
static __device__ inline v16bf load_b32x16(const bf16_t* base, int ld, int col0, int k0, int lane) {
  const int n  = lane & 15;
  const int hi = lane >> 4;
  const bf16_t* p = base + (size_t)(col0 + n) * ld + k0 + 8 * hi;
  v8bf x0 = *(const v8bf*)(p);
  v8bf x1 = *(const v8bf*)(p + 16);
  return pack16(x0, x1);
}

// ---------------------------------------------------------------------------
// Prep kernels: f32 -> bf16 convert, and LoRA fold W_eff = W + (alpha/r)*B@A
// ---------------------------------------------------------------------------
__global__ __launch_bounds__(256) void cpt_cvt_bf16(const float* __restrict__ x,
                                                    bf16_t* __restrict__ y, int n) {
  int i = blockIdx.x * 256 + threadIdx.x;
  if (i < n) y[i] = (bf16_t)x[i];
}

__global__ __launch_bounds__(256) void cpt_lora_fold(const float* __restrict__ W,
                                                     const float* __restrict__ A,
                                                     const float* __restrict__ Bm,
                                                     bf16_t* __restrict__ out, int N, int K) {
  int i = blockIdx.x * 256 + threadIdx.x;
  if (i >= N * K) return;
  int n = i / K;
  int k = i - n * K;
  float acc = W[i];
#pragma unroll
  for (int r = 0; r < RANK; ++r)
    acc += 2.0f * Bm[n * RANK + r] * A[r * K + k];   // ALPHA/RANK = 32/16 = 2
  out[i] = (bf16_t)acc;
}

// ---------------------------------------------------------------------------
// Software-pipelined GEMM core: wave computes a 16x64 tile of X(16xK) * W^T.
// Operands for step k+32 are loaded into fresh registers before the WMMAs for
// step k are issued, so loads overlap the matrix pipe (partial loadcnt waits
// instead of full drains).
// ---------------------------------------------------------------------------
static __device__ inline void gemm_tile_16x64(const bf16_t* __restrict__ X,
                                              const bf16_t* __restrict__ W,
                                              int row0, int n0, int lane,
                                              v8f acc[4]) {
  v16bf a_cur = load_a16x32(X, CDIM, row0, 0, lane);
  v16bf b_cur[4];
#pragma unroll
  for (int j = 0; j < 4; ++j) b_cur[j] = load_b32x16(W, CDIM, n0 + 16 * j, 0, lane);

  for (int k0 = 32; k0 < CDIM; k0 += 32) {
    // Prefetch next slice into fresh registers (no dependency on the WMMAs).
    v16bf a_nxt = load_a16x32(X, CDIM, row0, k0, lane);
    v16bf b_nxt[4];
#pragma unroll
    for (int j = 0; j < 4; ++j) b_nxt[j] = load_b32x16(W, CDIM, n0 + 16 * j, k0, lane);

#pragma unroll
    for (int j = 0; j < 4; ++j) acc[j] = WMMA_BF16(a_cur, b_cur[j], acc[j]);

    a_cur = a_nxt;
#pragma unroll
    for (int j = 0; j < 4; ++j) b_cur[j] = b_nxt[j];
  }
#pragma unroll
  for (int j = 0; j < 4; ++j) acc[j] = WMMA_BF16(a_cur, b_cur[j], acc[j]);
}

// ---------------------------------------------------------------------------
// QKV GEMM: [4096 x 1024] x [3072 x 1024]^T, wave computes 16x64 tile.
// Epilogue: bias add, split q/k/v, fake-quant k/v, scale q by 1/sqrt(64),
// store q/k row-major per head, v transposed [d][t].
// ---------------------------------------------------------------------------
__global__ __launch_bounds__(256) void cpt_qkv_gemm(const bf16_t* __restrict__ X,
                                                    const bf16_t* __restrict__ W,
                                                    const float* __restrict__ bias,
                                                    const float* __restrict__ kv_scale,
                                                    const float* __restrict__ kv_zp,
                                                    bf16_t* __restrict__ qbuf,
                                                    bf16_t* __restrict__ kbuf,
                                                    bf16_t* __restrict__ vT) {
  const int lane = threadIdx.x & 31;
  const int wid  = blockIdx.x * 8 + (threadIdx.x >> 5);
  const int NT   = NQKV / 64;          // 48 col-tiles
  const int mt   = wid / NT;
  const int nt   = wid - mt * NT;
  const int row0 = mt * 16;
  const int n0   = nt * 64;

  v8f acc[4];
#pragma unroll
  for (int j = 0; j < 4; ++j) acc[j] = vzero8();

  gemm_tile_16x64(X, W, row0, n0, lane, acc);

  const float scale = kv_scale[0];
  const float zp    = kv_zp[0];
  const int hi = lane >> 4;
  const int nc = lane & 15;
#pragma unroll
  for (int j = 0; j < 4; ++j) {
    const int   n    = n0 + 16 * j + nc;
    const float bcol = bias[n];
    const int   c    = n & (CDIM - 1);
    const int   hh   = c >> 6;
    const int   d    = c & 63;
#pragma unroll
    for (int r = 0; r < 8; ++r) {
      const int m  = row0 + r + 8 * hi;
      const int bb = m >> 11;            // / SEQ
      const int t  = m & (SEQ - 1);
      const size_t hidx = (((size_t)bb * NHEAD + hh) * SEQ + t) * HDIM + d;
      float val = acc[j][r] + bcol;
      if (n < CDIM) {
        qbuf[hidx] = (bf16_t)(val * 0.125f);       // fold 1/sqrt(HDIM)
      } else {
        float q = rintf(val / scale + zp);         // RTE like jnp.round
        q = fminf(fmaxf(q, 0.0f), 255.0f);
        float dq = (q - zp) * scale;
        if (n < 2 * CDIM) kbuf[hidx] = (bf16_t)dq;
        else vT[(((size_t)bb * NHEAD + hh) * HDIM + d) * SEQ + t] = (bf16_t)dq;
      }
    }
  }
}

// ---------------------------------------------------------------------------
// Causal flash attention: one wave per 16-query tile per (b,h).
// KV tiles of width 32. Pipelining: next tile's K fragments and current
// tile's V fragments are loaded before the softmax VALU block, so global
// latency hides behind exp2/shuffle work and the LDS P hand-off.
// ---------------------------------------------------------------------------
__global__ __launch_bounds__(256) void cpt_flash_attn(const bf16_t* __restrict__ qbuf,
                                                      const bf16_t* __restrict__ kbuf,
                                                      const bf16_t* __restrict__ vT,
                                                      bf16_t* __restrict__ ctx) {
  __shared__ __align__(16) bf16_t pbuf[8][16][32];   // per-wave P tile
  const int lane = threadIdx.x & 31;
  const int wib  = threadIdx.x >> 5;
  const int wid  = blockIdx.x * 8 + wib;
  const int qt = wid & (SEQ / 16 - 1);
  const int hh = (wid >> 7) & (NHEAD - 1);
  const int bb = wid >> 11;
  const int q0 = qt * 16;
  const int hi = lane >> 4;
  const int nc = lane & 15;

  const bf16_t* qh = qbuf + ((size_t)bb * NHEAD + hh) * SEQ * HDIM;
  const bf16_t* kh = kbuf + ((size_t)bb * NHEAD + hh) * SEQ * HDIM;
  const bf16_t* vh = vT   + ((size_t)bb * NHEAD + hh) * HDIM * SEQ;

  const v16bf qa0 = load_a16x32(qh, HDIM, q0, 0,  lane);
  const v16bf qa1 = load_a16x32(qh, HDIM, q0, 32, lane);

  v8f o[4];
#pragma unroll
  for (int j = 0; j < 4; ++j) o[j] = vzero8();
  float mst[8], lst[8];
#pragma unroll
  for (int r = 0; r < 8; ++r) { mst[r] = -3.0e38f; lst[r] = 0.0f; }

  const int ntile = (q0 + 47) >> 5;        // 32-wide kv tiles covering [0, q0+15]

  // Preload K fragments for the first kv tile.
  v16bf kb[4];
  kb[0] = load_b32x16(kh, HDIM, 0,  0,  lane);
  kb[1] = load_b32x16(kh, HDIM, 0,  32, lane);
  kb[2] = load_b32x16(kh, HDIM, 16, 0,  lane);
  kb[3] = load_b32x16(kh, HDIM, 16, 32, lane);

  for (int it = 0; it < ntile; ++it) {
    const int t0 = it * 32;
    v8f s0 = vzero8(), s1 = vzero8();
    s0 = WMMA_BF16(qa0, kb[0], s0);
    s0 = WMMA_BF16(qa1, kb[1], s0);
    s1 = WMMA_BF16(qa0, kb[2], s1);
    s1 = WMMA_BF16(qa1, kb[3], s1);

    // Prefetch V fragments for THIS tile (consumed after softmax) and K
    // fragments for the NEXT tile (consumed next iteration) — both overlap
    // the softmax VALU/transcendental block below.
    v16bf vb[4];
#pragma unroll
    for (int j = 0; j < 4; ++j) vb[j] = load_b32x16(vh, SEQ, j * 16, t0, lane);
    if (it + 1 < ntile) {
      const int tn = t0 + 32;
      kb[0] = load_b32x16(kh, HDIM, tn,      0,  lane);
      kb[1] = load_b32x16(kh, HDIM, tn,      32, lane);
      kb[2] = load_b32x16(kh, HDIM, tn + 16, 0,  lane);
      kb[3] = load_b32x16(kh, HDIM, tn + 16, 32, lane);
    }

    const int c0 = t0 + nc;
    const int c1 = c0 + 16;
#pragma unroll
    for (int r = 0; r < 8; ++r) {
      const int row = q0 + r + 8 * hi;
      float a0 = (c0 <= row) ? s0[r] : -3.0e38f;
      float a1 = (c1 <= row) ? s1[r] : -3.0e38f;
      float mx = fmaxf(a0, a1);
#pragma unroll
      for (int off = 8; off >= 1; off >>= 1) mx = fmaxf(mx, __shfl_xor(mx, off, 32));
      const float mnew = fmaxf(mst[r], mx);
      const float alph = exp2f((mst[r] - mnew) * 1.44269504f);
      const float p0   = exp2f((a0 - mnew) * 1.44269504f);
      const float p1   = exp2f((a1 - mnew) * 1.44269504f);
      float rs = p0 + p1;
#pragma unroll
      for (int off = 8; off >= 1; off >>= 1) rs += __shfl_xor(rs, off, 32);
      lst[r] = lst[r] * alph + rs;
      mst[r] = mnew;
#pragma unroll
      for (int j = 0; j < 4; ++j) o[j][r] *= alph;
      pbuf[wib][r + 8 * hi][nc]      = (bf16_t)p0;
      pbuf[wib][r + 8 * hi][nc + 16] = (bf16_t)p1;
    }
    // Cross-lane LDS hand-off within the wave: wait on the split DS counter.
    asm volatile("s_wait_dscnt 0" ::: "memory");
    v8bf x0 = *(const v8bf*)&pbuf[wib][nc][8 * hi];
    v8bf x1 = *(const v8bf*)&pbuf[wib][nc][16 + 8 * hi];
    const v16bf pa = pack16(x0, x1);
#pragma unroll
    for (int j = 0; j < 4; ++j) o[j] = WMMA_BF16(pa, vb[j], o[j]);
    asm volatile("" ::: "memory");   // keep LDS reads ordered vs next-iter stores
  }

#pragma unroll
  for (int r = 0; r < 8; ++r) lst[r] = 1.0f / lst[r];
#pragma unroll
  for (int j = 0; j < 4; ++j) {
    const int d = j * 16 + nc;
#pragma unroll
    for (int r = 0; r < 8; ++r) {
      const int row = q0 + r + 8 * hi;
      ctx[((size_t)bb * SEQ + row) * CDIM + hh * HDIM + d] = (bf16_t)(o[j][r] * lst[r]);
    }
  }
}

// ---------------------------------------------------------------------------
// Output projection GEMM: [4096 x 1024] x [1024 x 1024]^T -> f32 d_out.
// ---------------------------------------------------------------------------
__global__ __launch_bounds__(256) void cpt_proj_gemm(const bf16_t* __restrict__ ctx,
                                                     const bf16_t* __restrict__ W,
                                                     const float* __restrict__ bias,
                                                     float* __restrict__ out) {
  const int lane = threadIdx.x & 31;
  const int wid  = blockIdx.x * 8 + (threadIdx.x >> 5);
  const int NT   = CDIM / 64;   // 16
  const int mt   = wid / NT;
  const int nt   = wid - mt * NT;
  const int row0 = mt * 16;
  const int n0   = nt * 64;

  v8f acc[4];
#pragma unroll
  for (int j = 0; j < 4; ++j) acc[j] = vzero8();

  gemm_tile_16x64(ctx, W, row0, n0, lane, acc);

  const int hi = lane >> 4;
  const int nc = lane & 15;
#pragma unroll
  for (int j = 0; j < 4; ++j) {
    const int   n  = n0 + 16 * j + nc;
    const float bc = bias[n];
#pragma unroll
    for (int r = 0; r < 8; ++r) {
      const int m = row0 + r + 8 * hi;
      out[(size_t)m * CDIM + n] = acc[j][r] + bc;
    }
  }
}

// ---------------------------------------------------------------------------
// Host launcher
// ---------------------------------------------------------------------------
extern "C" void kernel_launch(void* const* d_in, const int* in_sizes, int n_in,
                              void* d_out, int out_size, void* d_ws, size_t ws_size,
                              hipStream_t stream) {
  (void)in_sizes; (void)n_in; (void)out_size; (void)ws_size;
  const float* hs       = (const float*)d_in[0];
  const float* W_attn   = (const float*)d_in[1];
  const float* b_attn   = (const float*)d_in[2];
  const float* A_attn   = (const float*)d_in[3];
  const float* B_attn   = (const float*)d_in[4];
  const float* W_proj   = (const float*)d_in[5];
  const float* b_proj   = (const float*)d_in[6];
  const float* A_proj   = (const float*)d_in[7];
  const float* B_proj   = (const float*)d_in[8];
  const float* kv_scale = (const float*)d_in[9];
  const float* kv_zp    = (const float*)d_in[10];

  char* p = (char*)d_ws;
  bf16_t* Xbf = (bf16_t*)p; p += (size_t)MROWS * CDIM * 2;   // 8 MiB
  bf16_t* Wq  = (bf16_t*)p; p += (size_t)NQKV  * CDIM * 2;   // 6 MiB
  bf16_t* Wp  = (bf16_t*)p; p += (size_t)CDIM  * CDIM * 2;   // 2 MiB
  bf16_t* qb  = (bf16_t*)p; p += (size_t)MROWS * CDIM * 2;   // 8 MiB
  bf16_t* kb  = (bf16_t*)p; p += (size_t)MROWS * CDIM * 2;   // 8 MiB
  bf16_t* vt  = (bf16_t*)p; p += (size_t)MROWS * CDIM * 2;   // 8 MiB
  bf16_t* ctx = (bf16_t*)p; p += (size_t)MROWS * CDIM * 2;   // 8 MiB  (~48 MiB total)

  cpt_cvt_bf16 <<<(MROWS * CDIM) / 256, 256, 0, stream>>>(hs, Xbf, MROWS * CDIM);
  cpt_lora_fold<<<(NQKV  * CDIM) / 256, 256, 0, stream>>>(W_attn, A_attn, B_attn, Wq, NQKV, CDIM);
  cpt_lora_fold<<<(CDIM  * CDIM) / 256, 256, 0, stream>>>(W_proj, A_proj, B_proj, Wp, CDIM, CDIM);

  // (4096/16) * (3072/64) = 12288 waves -> 1536 blocks of 8 waves
  cpt_qkv_gemm<<<(MROWS / 16) * (NQKV / 64) / 8, 256, 0, stream>>>(
      Xbf, Wq, b_attn, kv_scale, kv_zp, qb, kb, vt);

  // B*H*(T/16) = 4096 waves -> 512 blocks
  cpt_flash_attn<<<(BATCH * NHEAD * (SEQ / 16)) / 8, 256, 0, stream>>>(qb, kb, vt, ctx);

  // (4096/16) * (1024/64) = 4096 waves -> 512 blocks
  cpt_proj_gemm<<<(MROWS / 16) * (CDIM / 64) / 8, 256, 0, stream>>>(
      ctx, Wp, b_proj, (float*)d_out);
}
